// PQDenseLayer_6425271075127
// MI455X (gfx1250) — matrix operand order repaired and checked by reading, not compiled, not run on hardware
//
#include <hip/hip_runtime.h>

typedef __attribute__((ext_vector_type(16))) _Float16 v16h;
typedef __attribute__((ext_vector_type(8)))  _Float16 v8h;
typedef __attribute__((ext_vector_type(8)))  float    v8f;
typedef __attribute__((ext_vector_type(4)))  float    v4f;

#define B_N      8192
#define D_IN     4096
#define D_OUT    4096
#define M_CH     32
#define KS       256
#define CH       128

#define BT       16          // batch rows per workgroup (one WMMA M tile)
#define DT       2048        // output cols per workgroup
#define NWAVES   16
#define NTHREADS (NWAVES * 32)

#define CW       64          // c-half width staged in LDS per pass
#define STW      72          // f16 stride of sW rows: 144B, 16B-aligned, gcd(36,64)=4 banks
#define STP      264         // f32 stride of proj rows (256 + 8 pad)

// Static LDS: 256*72*2 + 16*264*4 = 36864 + 16896 = 53760 bytes (< 64KB)
__shared__ _Float16 sW[KS * STW];     // W[m] half-chunk, transposed: [n][c_local]
__shared__ float    sP[BT * STP];     // proj tile (includes bias): [row][ks]

extern "C" __global__ __launch_bounds__(NTHREADS)
void pq_fused_kernel(const float* __restrict__ x,     // (B, D_IN)
                     const float* __restrict__ W,     // (M, CH, KS)
                     const float* __restrict__ bias,  // (M, KS)
                     const int*   __restrict__ idx,   // (M, D_OUT)
                     float*       __restrict__ out)   // (B, D_OUT)
{
    const int tid  = threadIdx.x;
    const int lane = tid & 31;
    const int wv   = tid >> 5;        // 0..15
    const int hi   = lane >> 4;       // 0/1: which K-half of the fragment
    const int ln   = lane & 15;       // row (A/C) or column (B/C) index

    const int b0 = blockIdx.x * BT;   // batch tile base
    const int d0 = blockIdx.y * DT;   // d_out tile base
    const int n0 = wv * 16;           // ks column tile owned by this wave

    // acc[r][j] -> out[b0+r][d0 + wv*128 + lane + 32*j]
    float acc[BT][4];
#pragma unroll
    for (int r = 0; r < BT; ++r)
#pragma unroll
        for (int j = 0; j < 4; ++j) acc[r][j] = 0.0f;

    const float* xrow = x + (size_t)(b0 + ln) * D_IN;

    for (int m = 0; m < M_CH; ++m) {
        const float* Wm = W + (size_t)m * (CH * KS);
        const float* xm = xrow + m * CH;

        v8f c8 = {};                  // 16x16 f32 proj sub-tile accumulator

#pragma unroll
        for (int h = 0; h < 2; ++h) {
            // ---- stage W[m][h*64 .. h*64+63][:] into LDS as f16, transposed ----
            const float* Wh = Wm + h * (CW * KS);
#pragma unroll 4
            for (int i = tid; i < CW * KS; i += NTHREADS) {
                // i = c_local*256 + n  (contiguous global read, transposed LDS write)
                sW[(i & (KS - 1)) * STW + (i >> 8)] = (_Float16)Wh[i];
            }
            if (h == 0 && m + 1 < M_CH) {
                // pull next chunk's W toward L2 while we compute
                __builtin_prefetch(W + (size_t)(m + 1) * (CH * KS) + tid * 64, 0, 1);
            }
            __syncthreads();

#pragma unroll
            for (int kq = 0; kq < 2; ++kq) {
                const int kk = h * 2 + kq;      // global K step (32 each)

                // A fragment (16x32 f16): lane row = ln; K runs per ISA layout:
                // lanes 0-15: [32kk+0..7],[32kk+16..23]; lanes 16-31: +8 on each run
                v4f a0 = *(const v4f*)(xm + kk * 32 + hi * 8);
                v4f a1 = *(const v4f*)(xm + kk * 32 + hi * 8 + 4);
                v4f a2 = *(const v4f*)(xm + kk * 32 + 16 + hi * 8);
                v4f a3 = *(const v4f*)(xm + kk * 32 + 16 + hi * 8 + 4);
                v16h a;
#pragma unroll
                for (int e = 0; e < 4; ++e) {
                    a[e]      = (_Float16)a0[e];
                    a[4 + e]  = (_Float16)a1[e];
                    a[8 + e]  = (_Float16)a2[e];
                    a[12 + e] = (_Float16)a3[e];
                }

                // B fragment (32x16 f16): lane col = n0+ln, c runs [hi*8..+7],[16+hi*8..+7]
                const _Float16* bp = sW + (n0 + ln) * STW + kq * 32 + hi * 8;
                v8h bA = *(const v8h*)bp;         // ds_load_b128
                v8h bB = *(const v8h*)(bp + 16);  // ds_load_b128
                v16h bf = __builtin_shufflevector(bA, bB,
                          0, 1, 2, 3, 4, 5, 6, 7, 8, 9, 10, 11, 12, 13, 14, 15);

                c8 = __builtin_amdgcn_wmma_f32_16x16x32_f16(
                        false, a, false, bf, (short)0, c8, false, false);
            }
            __syncthreads();          // done reading this W half before overwrite
        }

        // ---- bias + store proj sub-tile (C layout: row = v + 8*hi, col = n0+ln) ----
        const float bv = bias[m * KS + n0 + ln];
#pragma unroll
        for (int v = 0; v < 8; ++v) {
            sP[(v + 8 * hi) * STP + n0 + ln] = c8[v] + bv;
        }
        __syncthreads();              // proj tile visible to all waves

        // ---- gather-accumulate: acc[r][j] += proj[r][idx[m][d]] ----
        const int* ix = idx + (size_t)m * D_OUT + d0 + wv * 128 + lane;
        const int k0 = ix[0], k1 = ix[32], k2 = ix[64], k3 = ix[96];
#pragma unroll
        for (int r = 0; r < BT; ++r) {
            const float* pr = sP + r * STP;
            acc[r][0] += pr[k0];
            acc[r][1] += pr[k1];
            acc[r][2] += pr[k2];
            acc[r][3] += pr[k3];
        }
        // no barrier needed here: next iteration's first sW write touches a
        // disjoint LDS region, and its following barrier orders sP reuse.
    }

    // ---- write out tile (coalesced, stride-32 within wave) ----
#pragma unroll
    for (int r = 0; r < BT; ++r) {
        float* orow = out + (size_t)(b0 + r) * D_OUT + d0 + wv * 128 + lane;
        orow[0]  = acc[r][0];
        orow[32] = acc[r][1];
        orow[64] = acc[r][2];
        orow[96] = acc[r][3];
    }
}

extern "C" void kernel_launch(void* const* d_in, const int* in_sizes, int n_in,
                              void* d_out, int out_size, void* d_ws, size_t ws_size,
                              hipStream_t stream) {
    const float* x    = (const float*)d_in[0];
    const float* W    = (const float*)d_in[1];
    const float* bias = (const float*)d_in[2];
    const int*   idx  = (const int*)d_in[3];
    float*       out  = (float*)d_out;

    dim3 grid(B_N / BT, D_OUT / DT, 1);     // 512 x 2
    dim3 block(NTHREADS, 1, 1);             // 512 threads = 16 waves
    pq_fused_kernel<<<grid, block, 0, stream>>>(x, W, bias, idx, out);
}
